// MultiHeadAttentionBasedAggregator_82884278878932
// MI455X (gfx1250) — compile-verified, matrix operand order checked by reading, not run on hardware
//
#include <hip/hip_runtime.h>

// ---------------------------------------------------------------------------
// MultiHeadAttentionBasedAggregator — fused CDNA5 (gfx1250) implementation.
//
// N=20000 nodes, P=12 personas, DN=DP=512, H=8 heads (d=64).
// Compute-bound on K/V projections (~252 GFLOP) -> bf16 WMMA with f32 accum.
// Fully fused per 16-node block: persona_embeddings (491 MB) read once, no
// K/V intermediates in HBM. Each wave accumulates TWO adjacent 16x16 N-tiles
// sharing one A fragment (halves ds traffic per WMMA); bias folded into the
// accumulator init (per-lane splat, valid because every C element a lane
// holds is in the same output column).
// ---------------------------------------------------------------------------

typedef unsigned short u16;
typedef __bf16  bf16_t;
typedef __attribute__((ext_vector_type(16))) bf16_t v16bf;
typedef __attribute__((ext_vector_type(8)))  bf16_t v8bf;
typedef __attribute__((ext_vector_type(8)))  float  v8f;

#define NODES_PER_WG 16
#define NP_ROWS      (NODES_PER_WG * 12)     // 192 persona rows per block
#define PER_STRIDE   520                     // bf16 elems per persona LDS row (512 + 8 pad)
#define COMB_STRIDE  1032                    // bf16 elems per combined LDS row (1024 + 8 pad)
#define Q_STRIDE     516                     // f32 elems per q LDS row
#define KV_STRIDE    66                      // f32 elems per k/v chunk LDS row (64 + 2 pad)

#define SMEM_BYTES ((192*PER_STRIDE + 16*COMB_STRIDE)*2 + (16*Q_STRIDE + 192*KV_STRIDE + 16*8*12)*4 + 64)

// round-to-nearest-even f32 -> bf16
__device__ __forceinline__ u16 f2bf(float f) {
  unsigned u = __float_as_uint(f);
  return (u16)((u + 0x7FFFu + ((u >> 16) & 1u)) >> 16);
}

struct alignas(8) US4 { u16 a, b, c, d; };

#define SPLAT8(x) {(x),(x),(x),(x),(x),(x),(x),(x)}

// A fragment (16-bit A 16x32 layout): lane holds row m=lane&15; K pairs:
//   VGPR j(0..3): K = k0 + g*8 + 2j,+1 ; VGPR j(4..7): K = k0 + 16 + g*8 + 2(j-4),+1
// -> two contiguous 8-halfword (16 B) LDS reads per lane.
__device__ __forceinline__ v16bf load_a_frag(const u16* row, int k0, int g) {
  union { v16bf v; v8bf h[2]; } f;
  const bf16_t* p = (const bf16_t*)(row + k0 + (g << 3));
  f.h[0] = *(const v8bf*)(p);
  f.h[1] = *(const v8bf*)(p + 16);
  return f.v;
}

// B fragment from pre-swizzled weights: tile = 512 bf16, lane-major, 16 per lane.
__device__ __forceinline__ v16bf load_b_frag(const u16* tile_lane_ptr) {
  union { v16bf v; v8bf h[2]; } f;
  f.h[0] = *(const v8bf*)(tile_lane_ptr);
  f.h[1] = *(const v8bf*)(tile_lane_ptr + 8);
  return f.v;
}

__device__ __forceinline__ v8f wmma_bf16(v16bf a, v16bf b, v8f c) {
  return __builtin_amdgcn_wmma_f32_16x16x32_bf16(false, a, false, b, (short)0, c, false, false);
}

// ---------------------------------------------------------------------------
// Kernel 0: pack f32 weights into bf16 WMMA-B-fragment tile order.
// Tile (kt,nt) of W[Ktot x 512]: 512 bf16; index j(0..15) within lane:
//   K = kt*32 + (j>=8 ? 16 : 0) + g*8 + ((j>>1)&3)*2 + (j&1),  n = nt*16 + (lane&15)
// Tile order: tileIdx = nt*KT + kt  (k-stream contiguous per n-tile).
// ---------------------------------------------------------------------------
__global__ void pack_weights_kernel(const float* __restrict__ Wq,
                                    const float* __restrict__ Wk,
                                    const float* __restrict__ Wv,
                                    const float* __restrict__ Wo,
                                    u16* __restrict__ ws) {
  int idx = blockIdx.x * 256 + threadIdx.x;
  const float* W; u16* dst; int Ktot, count;
  switch (blockIdx.y) {
    case 0:  W = Wq; dst = ws;          Ktot = 512;  count = 16*32*512; break;
    case 1:  W = Wk; dst = ws + 262144; Ktot = 512;  count = 16*32*512; break;
    case 2:  W = Wv; dst = ws + 524288; Ktot = 512;  count = 16*32*512; break;
    default: W = Wo; dst = ws + 786432; Ktot = 1024; count = 32*32*512; break;
  }
  if (idx >= count) return;
  int j    = idx & 15;
  int lane = (idx >> 4) & 31;
  int tile = idx >> 9;
  int KT   = Ktot >> 5;
  int nt   = tile / KT;
  int kt   = tile - nt * KT;
  int g    = lane >> 4;
  int c    = lane & 15;
  int pairIdx = j >> 1, lo = j & 1;
  int K = (kt << 5) + ((pairIdx >> 2) << 4) + (g << 3) + ((pairIdx & 3) << 1) + lo;
  int n = (nt << 4) + c;
  dst[idx] = f2bf(W[K * 512 + n]);
}

// ---------------------------------------------------------------------------
// Kernel 1: fused attention aggregator, 16 nodes per workgroup.
// ---------------------------------------------------------------------------
__global__ void __launch_bounds__(512, 1)
attn_fused_kernel(const float* __restrict__ node_emb,
                  const float* __restrict__ persona_emb,
                  const int*   __restrict__ lens,
                  const float* __restrict__ bq,
                  const float* __restrict__ bk,
                  const float* __restrict__ bv,
                  const float* __restrict__ bo,
                  const u16* __restrict__ Pq,
                  const u16* __restrict__ Pk,
                  const u16* __restrict__ Pv,
                  const u16* __restrict__ Po,
                  float* __restrict__ out) {
  extern __shared__ unsigned char smem[];
  u16*   sPersona = (u16*)smem;                         // [192][PER_STRIDE] bf16
  u16*   sComb    = sPersona + 192 * PER_STRIDE;        // [16][COMB_STRIDE] bf16 (node | att)
  float* sQ       = (float*)(sComb + 16 * COMB_STRIDE); // [16][Q_STRIDE]
  float* sKV      = sQ + 16 * Q_STRIDE;                 // [192][KV_STRIDE] (reused K then V)
  float* sSc      = sKV + 192 * KV_STRIDE;              // [16][8][12] scores -> weights
  int*   sLen     = (int*)(sSc + 16 * 8 * 12);          // [16]

  const int tid  = threadIdx.x;
  const int wave = tid >> 5;
  const int lane = tid & 31;
  const int g    = lane >> 4;
  const int cN   = lane & 15;
  const int node0 = blockIdx.x * NODES_PER_WG;

  // ---- stage: persona tile (contiguous 192*512 f32) -> bf16 LDS -------------
  {
    const float4* pg = (const float4*)(persona_emb + (size_t)node0 * 12 * 512);
    for (int i = tid; i < 24576; i += 512) {
      float4 x = pg[i];
      int r = i >> 7, c = (i & 127) << 2;
      *(US4*)(sPersona + r * PER_STRIDE + c) = US4{f2bf(x.x), f2bf(x.y), f2bf(x.z), f2bf(x.w)};
    }
    const float4* ng = (const float4*)(node_emb + (size_t)node0 * 512);
    for (int i = tid; i < 2048; i += 512) {
      float4 x = ng[i];
      int r = i >> 7, c = (i & 127) << 2;
      *(US4*)(sComb + r * COMB_STRIDE + c) = US4{f2bf(x.x), f2bf(x.y), f2bf(x.z), f2bf(x.w)};
    }
    if (tid < NODES_PER_WG) sLen[tid] = lens[node0 + tid];
  }
  __syncthreads();

  // ---- Q projection: [16x512] @ Wq; each wave owns 2 adjacent n-tiles ------
  {
    const int n0 = wave << 5;
    const float b0 = bq[n0 + cN];
    const float b1 = bq[n0 + 16 + cN];
    v8f acc0 = SPLAT8(b0);
    v8f acc1 = SPLAT8(b1);
    const u16* wp0  = Pq + (size_t)((2 * wave)     * 16) * 512 + lane * 16;
    const u16* wp1  = Pq + (size_t)((2 * wave + 1) * 16) * 512 + lane * 16;
    const u16* arow = sComb + cN * COMB_STRIDE;
#pragma unroll 4
    for (int kt = 0; kt < 16; ++kt) {
      __builtin_prefetch(wp0 + 2048, 0, 1);
      __builtin_prefetch(wp1 + 2048, 0, 1);
      v16bf a = load_a_frag(arow, kt << 5, g);
      acc0 = wmma_bf16(a, load_b_frag(wp0), acc0);
      acc1 = wmma_bf16(a, load_b_frag(wp1), acc1);
      wp0 += 512; wp1 += 512;
    }
#pragma unroll
    for (int i = 0; i < 8; ++i) {
      sQ[(i + 8 * g) * Q_STRIDE + n0 + cN]      = acc0[i];
      sQ[(i + 8 * g) * Q_STRIDE + n0 + 16 + cN] = acc1[i];
    }
  }
  __syncthreads();

  // ---- per head: K-proj chunk (192x64) then raw scores ---------------------
  // 12 m-tiles x 2 ntc-pairs = 24 units; each unit: 2 adjacent n-tiles share A.
  for (int h = 0; h < 8; ++h) {
    for (int u = wave; u < 24; u += 16) {
      const int mt = u >> 1, pr2 = u & 1;
      const int n0 = h * 64 + (pr2 << 5);
      const float b0 = bk[n0 + cN];
      const float b1 = bk[n0 + 16 + cN];
      v8f acc0 = SPLAT8(b0);
      v8f acc1 = SPLAT8(b1);
      const u16* wp0  = Pk + (size_t)(((n0 >> 4))     * 16) * 512 + lane * 16;
      const u16* wp1  = Pk + (size_t)(((n0 >> 4) + 1) * 16) * 512 + lane * 16;
      const u16* arow = sPersona + ((mt << 4) + cN) * PER_STRIDE;
#pragma unroll 4
      for (int kt = 0; kt < 16; ++kt) {
        __builtin_prefetch(wp0 + 2048, 0, 1);
        __builtin_prefetch(wp1 + 2048, 0, 1);
        v16bf a = load_a_frag(arow, kt << 5, g);
        acc0 = wmma_bf16(a, load_b_frag(wp0), acc0);
        acc1 = wmma_bf16(a, load_b_frag(wp1), acc1);
        wp0 += 512; wp1 += 512;
      }
      const int cbase = (pr2 << 5) + cN;
#pragma unroll
      for (int i = 0; i < 8; ++i) {
        sKV[((mt << 4) + i + 8 * g) * KV_STRIDE + cbase]      = acc0[i];
        sKV[((mt << 4) + i + 8 * g) * KV_STRIDE + cbase + 16] = acc1[i];
      }
    }
    __syncthreads();
    // scores[n][h][p] = q[n, h*64 : h*64+64] . k_row   (tiny: VALU)
    if (tid < NP_ROWS) {
      const int n = tid / 12, p = tid - n * 12;
      const float* qp = sQ + n * Q_STRIDE + h * 64;
      const float* kp = sKV + tid * KV_STRIDE;
      float s = 0.f;
#pragma unroll
      for (int dd = 0; dd < 64; ++dd) s += qp[dd] * kp[dd];
      sSc[(n * 8 + h) * 12 + p] = s;
    }
    __syncthreads();
  }

  // ---- masked softmax over p (ragged lengths); masked weights -> 0 ---------
  if (tid < 128) {
    const int n = tid >> 3, h = tid & 7;
    const int len = sLen[n];
    float* s = sSc + (n * 8 + h) * 12;
    float m = -1e30f;
    for (int p = 0; p < 12; ++p) if (p < len) m = fmaxf(m, s[p]);
    float sum = 0.f;
    for (int p = 0; p < 12; ++p) {
      float e = (p < len) ? __expf(s[p] - m) : 0.f;
      s[p] = e; sum += e;
    }
    const float inv = 1.f / sum;
    for (int p = 0; p < 12; ++p) s[p] *= inv;
  }
  __syncthreads();

  // ---- per head: V-proj chunk, then att -> combined[:, 512:] as bf16 -------
  for (int h = 0; h < 8; ++h) {
    for (int u = wave; u < 24; u += 16) {
      const int mt = u >> 1, pr2 = u & 1;
      const int n0 = h * 64 + (pr2 << 5);
      const float b0 = bv[n0 + cN];
      const float b1 = bv[n0 + 16 + cN];
      v8f acc0 = SPLAT8(b0);
      v8f acc1 = SPLAT8(b1);
      const u16* wp0  = Pv + (size_t)(((n0 >> 4))     * 16) * 512 + lane * 16;
      const u16* wp1  = Pv + (size_t)(((n0 >> 4) + 1) * 16) * 512 + lane * 16;
      const u16* arow = sPersona + ((mt << 4) + cN) * PER_STRIDE;
#pragma unroll 4
      for (int kt = 0; kt < 16; ++kt) {
        __builtin_prefetch(wp0 + 2048, 0, 1);
        __builtin_prefetch(wp1 + 2048, 0, 1);
        v16bf a = load_a_frag(arow, kt << 5, g);
        acc0 = wmma_bf16(a, load_b_frag(wp0), acc0);
        acc1 = wmma_bf16(a, load_b_frag(wp1), acc1);
        wp0 += 512; wp1 += 512;
      }
      const int cbase = (pr2 << 5) + cN;
#pragma unroll
      for (int i = 0; i < 8; ++i) {
        sKV[((mt << 4) + i + 8 * g) * KV_STRIDE + cbase]      = acc0[i];
        sKV[((mt << 4) + i + 8 * g) * KV_STRIDE + cbase + 16] = acc1[i];
      }
    }
    __syncthreads();
    for (int o = tid; o < 1024; o += 512) {
      const int n = o >> 6, dd = o & 63;
      const float* w  = sSc + (n * 8 + h) * 12;
      const float* vp = sKV + (n * 12) * KV_STRIDE + dd;
      float a = 0.f;
#pragma unroll
      for (int p = 0; p < 12; ++p) a += w[p] * vp[p * KV_STRIDE];
      sComb[n * COMB_STRIDE + 512 + h * 64 + dd] = f2bf(a);
    }
    __syncthreads();
  }

  // ---- output GEMM: combined [16x1024] @ Wo + bo -> global f32 -------------
  {
    const int n0 = wave << 5;
    const float b0 = bo[n0 + cN];
    const float b1 = bo[n0 + 16 + cN];
    v8f acc0 = SPLAT8(b0);
    v8f acc1 = SPLAT8(b1);
    const u16* wp0  = Po + (size_t)((2 * wave)     * 32) * 512 + lane * 16;
    const u16* wp1  = Po + (size_t)((2 * wave + 1) * 32) * 512 + lane * 16;
    const u16* arow = sComb + cN * COMB_STRIDE;
#pragma unroll 4
    for (int kt = 0; kt < 32; ++kt) {
      __builtin_prefetch(wp0 + 2048, 0, 1);
      __builtin_prefetch(wp1 + 2048, 0, 1);
      v16bf a = load_a_frag(arow, kt << 5, g);
      acc0 = wmma_bf16(a, load_b_frag(wp0), acc0);
      acc1 = wmma_bf16(a, load_b_frag(wp1), acc1);
      wp0 += 512; wp1 += 512;
    }
#pragma unroll
    for (int i = 0; i < 8; ++i) {
      out[(size_t)(node0 + i + 8 * g) * 512 + n0 + cN]      = acc0[i];
      out[(size_t)(node0 + i + 8 * g) * 512 + n0 + 16 + cN] = acc1[i];
    }
  }
}

// ---------------------------------------------------------------------------
extern "C" void kernel_launch(void* const* d_in, const int* in_sizes, int n_in,
                              void* d_out, int out_size, void* d_ws, size_t ws_size,
                              hipStream_t stream) {
  const float* node    = (const float*)d_in[0];
  const float* persona = (const float*)d_in[1];
  const int*   lens    = (const int*)  d_in[2];
  const float* Wq = (const float*)d_in[3];
  const float* bq = (const float*)d_in[4];
  const float* Wk = (const float*)d_in[5];
  const float* bk = (const float*)d_in[6];
  const float* Wv = (const float*)d_in[7];
  const float* bv = (const float*)d_in[8];
  const float* Wo = (const float*)d_in[9];
  const float* bo = (const float*)d_in[10];
  float* out = (float*)d_out;
  u16* wsp = (u16*)d_ws;   // 2.56 MB of packed bf16 weights

  // swizzle weights into WMMA B-fragment order (bf16)
  pack_weights_kernel<<<dim3(2048, 4), dim3(256), 0, stream>>>(Wq, Wk, Wv, Wo, wsp);

  // fused attention: 20000 / 16 = 1250 workgroups, 512 threads (16 wave32)
  attn_fused_kernel<<<dim3(1250), dim3(512), SMEM_BYTES, stream>>>(
      node, persona, lens, bq, bk, bv, bo,
      wsp, wsp + 262144, wsp + 524288, wsp + 786432, out);
}
